// Attention_33852932227326
// MI455X (gfx1250) — matrix operand order repaired and checked by reading, not compiled, not run on hardware
//
#include <hip/hip_runtime.h>
#include <stdint.h>

// ---------------------------------------------------------------------------
// MI455X (gfx1250) causal multi-head attention, bf16 WMMA everywhere.
// ~70 GFLOP vs ~100MB traffic -> matrix-core bound; all matmuls run on
// v_wmma_f32_16x16x32_bf16 with f32 accumulation (wave32 layouts per
// CDNA5 ISA 7.12.2). Non-transposed global->LDS tile copies use
// GLOBAL_LOAD_ASYNC_TO_LDS_B128 (ASYNCcnt) + s_wait_asynccnt; next-stage
// global data is prefetched with global_prefetch_b8.
// ---------------------------------------------------------------------------

typedef unsigned short u16;
typedef __bf16 v16bf __attribute__((ext_vector_type(16)));
typedef float  v8f   __attribute__((ext_vector_type(8)));

union FragBF { v16bf v; uint4 u[2]; };

__device__ __forceinline__ u16 f2bf(float f) {
  unsigned int u = __float_as_uint(f);
  u += 0x7fffu + ((u >> 16) & 1u);   // round-to-nearest-even
  return (u16)(u >> 16);
}

// 16B global -> LDS async copy (no VGPR round trip, tracked by ASYNCcnt).
// LDS byte address = low 32 bits of the generic shared pointer (aperture map).
__device__ __forceinline__ void async_copy16(const u16* gsrc, u16* ldst) {
  unsigned lds_off = (unsigned)(size_t)(void*)ldst;
  asm volatile("global_load_async_to_lds_b128 %0, %1, off"
               :: "v"(lds_off), "v"(gsrc) : "memory");
}
__device__ __forceinline__ void wait_async0() {
  asm volatile("s_wait_asynccnt 0x0" ::: "memory");
}

// ---------------------------------------------------------------------------
__global__ void f32_to_bf16(const float* __restrict__ in, u16* __restrict__ out, int n) {
  for (int i = blockIdx.x * blockDim.x + threadIdx.x; i < n; i += gridDim.x * blockDim.x)
    out[i] = f2bf(in[i]);
}

// ---------------------------------------------------------------------------
// Tiled bf16 GEMM: C[M x N] = A[M x K] * B[K x N], f32 accum.
// Block tile 128x64, 8 waves (4x2), each wave 32x32 (2x2 WMMA tiles), BK=32.
// mode 0: scatter bf16 to head-major [b][h][n][64] with scale (QKV proj).
// mode 1: f32 row-major + bias (final out-projection).
#define BM 128
#define BN 64
#define BK 32

__global__ __launch_bounds__(256) void gemm_bf16(
    const u16* __restrict__ A, int lda,
    const u16* __restrict__ B, int ldb,
    int N, int Ktot, float scale, int mode,
    u16* __restrict__ outb, float* __restrict__ outf, const float* __restrict__ bias)
{
  __shared__ __align__(16) u16 As[BM][BK];   // row-major [m][k]
  __shared__ __align__(16) u16 Bst[BN][BK];  // transposed [n][k]

  const int tid  = threadIdx.x;
  const int lane = tid & 31, wave = tid >> 5;
  const int wm = wave & 3, wn = wave >> 2;
  const int bx = blockIdx.x, by = blockIdx.y;
  const int m16 = lane & 15, h = lane >> 4;

  v8f acc[2][2] = {};

  for (int k0 = 0; k0 < Ktot; k0 += BK) {
    { // A tile: 128x32, two async 16B global->LDS copies per thread
      int ar = tid >> 2, ac = (tid & 3) * 8;
      const u16* ap = A + (size_t)(by * BM + ar) * lda + k0 + ac;
      async_copy16(ap,                    &As[ar][ac]);
      async_copy16(ap + (size_t)64 * lda, &As[ar + 64][ac]);
      if (k0 + BK < Ktot) __builtin_prefetch(ap + BK, 0, 1);  // next A stage
    }
    { // B tile: 32x64 loaded row-major, stored transposed (register transpose)
      int br = tid >> 3, bc = (tid & 7) * 8;
      uint4 t4 = *(const uint4*)(B + (size_t)(k0 + br) * ldb + bx * BN + bc);
      u16 tv[8]; *(uint4*)tv = t4;
      #pragma unroll
      for (int i = 0; i < 8; ++i) Bst[bc + i][br] = tv[i];
    }
    wait_async0();        // ASYNCcnt not covered by the barrier
    __syncthreads();

    FragBF fa[2], fb[2];
    #pragma unroll
    for (int mt = 0; mt < 2; ++mt) {       // A frag: row m, elems K = 8h+e / 16+8h+e
      int mrow = wm * 32 + mt * 16 + m16;
      fa[mt].u[0] = *(const uint4*)&As[mrow][h * 8];
      fa[mt].u[1] = *(const uint4*)&As[mrow][16 + h * 8];
    }
    #pragma unroll
    for (int nt = 0; nt < 2; ++nt) {       // B frag: col n, elems K = 8h+e / 16+8h+e
      int ncol = wn * 32 + nt * 16 + m16;
      fb[nt].u[0] = *(const uint4*)&Bst[ncol][h * 8];
      fb[nt].u[1] = *(const uint4*)&Bst[ncol][16 + h * 8];
    }
    #pragma unroll
    for (int mt = 0; mt < 2; ++mt)
      #pragma unroll
      for (int nt = 0; nt < 2; ++nt)
        acc[mt][nt] = __builtin_amdgcn_wmma_f32_16x16x32_bf16(
            false, fa[mt].v, false, fb[nt].v, (short)0, acc[mt][nt], false, false);
    __syncthreads();
  }

  // Epilogue: C/D layout lane n+16h, vgpr r -> row r+8h.
  #pragma unroll
  for (int mt = 0; mt < 2; ++mt) {
    #pragma unroll
    for (int nt = 0; nt < 2; ++nt) {
      int col = bx * BN + wn * 32 + nt * 16 + m16;
      #pragma unroll
      for (int r = 0; r < 8; ++r) {
        int row = by * BM + wm * 32 + mt * 16 + r + 8 * h;
        float v = acc[mt][nt][r] * scale;
        if (mode == 0) {
          int bi = row >> 11, ni = row & 2047;       // 2048 rows per batch
          int hh = col >> 6,  dd = col & 63;
          outb[((size_t)(bi * 16 + hh) * 2048 + ni) * 64 + dd] = f2bf(v);
        } else {
          outf[(size_t)row * N + col] = v + bias[col];
        }
      }
    }
  }
}

// ---------------------------------------------------------------------------
// Flash attention, causal. Grid: (n/128, b*heads). 8 waves, 16 rows each.
__global__ __launch_bounds__(256) void attn_fa(
    const u16* __restrict__ Qh, const u16* __restrict__ Kh,
    const u16* __restrict__ Vh, u16* __restrict__ Ob)
{
  __shared__ __align__(16) u16 Ks[32][64];     // K block row-major [key][d]
  __shared__ __align__(16) u16 Vts[64][32];    // V block transposed [d][key]
  __shared__ __align__(16) u16 Pl[8][16][32];  // per-wave P tile (C->A shuffle)

  const int tid  = threadIdx.x;
  const int lane = tid & 31, wave = tid >> 5;
  const int m16 = lane & 15, h = lane >> 4;
  const int rowBlock = blockIdx.x;
  const int bh = blockIdx.y;                   // bi*16 + head
  const int bi = bh >> 4, hi = bh & 15;
  const int i0 = rowBlock * 128 + wave * 16;   // first row owned by this wave

  const u16* Qb = Qh + (size_t)bh * 2048 * 64;
  const u16* Kb = Kh + (size_t)bh * 2048 * 64;
  const u16* Vb = Vh + (size_t)bh * 2048 * 64;

  // Q fragments (A layout) for d-chunks [0,32) and [32,64). Contiguous 16B loads.
  FragBF qa[2];
  {
    const u16* qr = Qb + (size_t)(i0 + m16) * 64;
    #pragma unroll
    for (int k = 0; k < 2; ++k) {
      qa[k].u[0] = *(const uint4*)&qr[k * 32 + h * 8];
      qa[k].u[1] = *(const uint4*)&qr[k * 32 + 16 + h * 8];
    }
  }

  v8f o[4] = {};
  float mrow[8], lrow[8];
  #pragma unroll
  for (int r = 0; r < 8; ++r) { mrow[r] = -3.0e38f; lrow[r] = 0.0f; }

  const int nblocks = (rowBlock + 1) * 4;      // 32-key blocks up to causal edge
  for (int jb = 0; jb < nblocks; ++jb) {
    const int j0 = jb * 32;
    __syncthreads();                            // previous block fully consumed
    {
      int kr = tid >> 3, c0 = (tid & 7) * 8;
      // K tile: straight copy -> async global->LDS (no VGPR round trip)
      async_copy16(&Kb[(size_t)(j0 + kr) * 64 + c0], &Ks[kr][c0]);
      // V tile: needs transpose -> load to regs, scatter b16 into LDS
      uint4 t4 = *(const uint4*)&Vb[(size_t)(j0 + kr) * 64 + c0];
      u16 tv[8]; *(uint4*)tv = t4;
      #pragma unroll
      for (int i = 0; i < 8; ++i) Vts[c0 + i][kr] = tv[i];
      if (jb + 1 < nblocks) {                  // prefetch next stage K/V
        __builtin_prefetch(&Kb[(size_t)(j0 + 32 + kr) * 64 + c0], 0, 1);
        __builtin_prefetch(&Vb[(size_t)(j0 + 32 + kr) * 64 + c0], 0, 1);
      }
    }
    wait_async0();
    __syncthreads();

    if (j0 <= i0 + 15) {                        // wave-uniform: EXEC all-ones for WMMA
      // S = Q * K^T  : two 16x16 tiles, K-dim = d (2 chunks of 32)
      v8f s[2] = {};
      #pragma unroll
      for (int jt = 0; jt < 2; ++jt) {
        int jc = jt * 16 + m16;                 // B frag: col = key, elems d = 32k+8h+e / 32k+16+8h+e
        #pragma unroll
        for (int k = 0; k < 2; ++k) {
          FragBF kb;
          kb.u[0] = *(const uint4*)&Ks[jc][k * 32 + h * 8];
          kb.u[1] = *(const uint4*)&Ks[jc][k * 32 + 16 + h * 8];
          s[jt] = __builtin_amdgcn_wmma_f32_16x16x32_bf16(
              false, qa[k].v, false, kb.v, (short)0, s[jt], false, false);
        }
      }
      // causal mask (only near the diagonal)
      if (j0 + 31 > i0) {
        #pragma unroll
        for (int jt = 0; jt < 2; ++jt) {
          int jc = j0 + jt * 16 + m16;
          #pragma unroll
          for (int r = 0; r < 8; ++r)
            if (jc > i0 + r + 8 * h) s[jt][r] = -3.0e38f;
        }
      }
      // online softmax: each row lives in 16 lanes of one half-wave
      #pragma unroll
      for (int r = 0; r < 8; ++r) {
        float mx = fmaxf(s[0][r], s[1][r]);
        #pragma unroll
        for (int dd = 1; dd < 16; dd <<= 1) mx = fmaxf(mx, __shfl_xor(mx, dd, 16));
        float mnew  = fmaxf(mrow[r], mx);
        float alpha = __expf(mrow[r] - mnew);
        float p0 = __expf(s[0][r] - mnew);
        float p1 = __expf(s[1][r] - mnew);
        float rs = p0 + p1;
        #pragma unroll
        for (int dd = 1; dd < 16; dd <<= 1) rs += __shfl_xor(rs, dd, 16);
        lrow[r] = lrow[r] * alpha + rs;
        mrow[r] = mnew;
        #pragma unroll
        for (int t = 0; t < 4; ++t) o[t][r] *= alpha;
        Pl[wave][r + 8 * h][m16]      = f2bf(p0);   // C layout -> LDS row-major
        Pl[wave][r + 8 * h][16 + m16] = f2bf(p1);
      }
      asm volatile("s_wait_dscnt 0" ::: "memory");  // cross-lane LDS visibility
      // O += P * V : A frag of P from LDS, B frags from transposed V
      FragBF pf;
      pf.u[0] = *(const uint4*)&Pl[wave][m16][h * 8];
      pf.u[1] = *(const uint4*)&Pl[wave][m16][16 + h * 8];
      #pragma unroll
      for (int t = 0; t < 4; ++t) {
        int dc = t * 16 + m16;
        FragBF vf;
        vf.u[0] = *(const uint4*)&Vts[dc][h * 8];
        vf.u[1] = *(const uint4*)&Vts[dc][16 + h * 8];
        o[t] = __builtin_amdgcn_wmma_f32_16x16x32_bf16(
            false, pf.v, false, vf.v, (short)0, o[t], false, false);
      }
    }
  }

  // normalize and store to [b][n][heads*64] bf16 for the out-projection GEMM
  #pragma unroll
  for (int t = 0; t < 4; ++t) {
    #pragma unroll
    for (int r = 0; r < 8; ++r) {
      int row = i0 + r + 8 * h;
      int col = hi * 64 + t * 16 + m16;
      float v = o[t][r] / lrow[r];
      Ob[(size_t)(bi * 2048 + row) * 1024 + col] = f2bf(v);
    }
  }
}

// ---------------------------------------------------------------------------
extern "C" void kernel_launch(void* const* d_in, const int* in_sizes, int n_in,
                              void* d_out, int out_size, void* d_ws, size_t ws_size,
                              hipStream_t stream)
{
  (void)in_sizes; (void)n_in; (void)out_size; (void)ws_size;
  const float* x     = (const float*)d_in[0];  // [2,2048,1024]
  const float* w_q   = (const float*)d_in[1];  // [1024,1024]
  const float* w_kv  = (const float*)d_in[2];  // [1024,2048]
  const float* w_out = (const float*)d_in[3];  // [1024,1024]
  const float* b_out = (const float*)d_in[4];  // [1024]
  float* out = (float*)d_out;                  // [2,2048,1024] f32

  const size_t M = 4096, D = 1024;
  // workspace layout (bf16 as u16), ~50MB total
  u16* xb   = (u16*)d_ws;          // 4096x1024
  u16* wqb  = xb   + M * D;        // 1024x1024
  u16* wkvb = wqb  + D * D;        // 1024x2048
  u16* wob  = wkvb + D * 2048;     // 1024x1024
  u16* qh   = wob  + D * D;        // [2][16][2048][64]
  u16* kh   = qh   + M * D;
  u16* vh   = kh   + M * D;
  u16* ob   = vh   + M * D;        // [2][2048][1024]

  f32_to_bf16<<<2048, 256, 0, stream>>>(x,     xb,   (int)(M * D));
  f32_to_bf16<<<1024, 256, 0, stream>>>(w_q,   wqb,  (int)(D * D));
  f32_to_bf16<<<2048, 256, 0, stream>>>(w_kv,  wkvb, (int)(D * 2048));
  f32_to_bf16<<<1024, 256, 0, stream>>>(w_out, wob,  (int)(D * D));

  dim3 g(1024 / BN, 4096 / BM);  // (16, 32)
  // Q gets scale*scale = 1/64 folded in (reference double-scales q)
  gemm_bf16<<<g, 256, 0, stream>>>(xb, 1024, wqb,         1024, 1024, 1024, 1.0f / 64.0f, 0, qh, nullptr, nullptr);
  gemm_bf16<<<g, 256, 0, stream>>>(xb, 1024, wkvb,        2048, 1024, 1024, 1.0f,         0, kh, nullptr, nullptr);
  gemm_bf16<<<g, 256, 0, stream>>>(xb, 1024, wkvb + 1024, 2048, 1024, 1024, 1.0f,         0, vh, nullptr, nullptr);

  attn_fa<<<dim3(16, 32), 256, 0, stream>>>(qh, kh, vh, ob);

  gemm_bf16<<<g, 256, 0, stream>>>(ob, 1024, wob, 1024, 1024, 1024, 1.0f, 1, nullptr, out, b_out);
}
